// Prop_54116587930003
// MI455X (gfx1250) — matrix operand order repaired and checked by reading, not compiled, not run on hardware
//
#include <hip/hip_runtime.h>
#include <hip/hip_fp16.h>
#include <math.h>

typedef __attribute__((ext_vector_type(16))) _Float16 v16h;
typedef __attribute__((ext_vector_type(8)))  _Float16 v8h;
typedef __attribute__((ext_vector_type(8)))  float    v8f;

namespace {
constexpr int kB   = 2;
constexpr int kCfi = 32;
constexpr int kN   = 192 * 192;   // 36864 spatial positions
constexpr int kNum = 16;
constexpr int kCt  = 67;          // 32 + 32 + 1 + 2 input channels of layer 1
constexpr int kK1  = 96;          // layer-1 K padded to 3 x 32
constexpr int kNT  = 256;         // tile columns = 16 num x 16 spatial
constexpr int kTN  = 16;          // spatial positions per block

// ---- LDS carve (element offsets) ----
constexpr int oX   = 0;                  // [kNT][kK1] f16  activations layer1 in
constexpr int oH1  = oX  + kNT * kK1;    // [kNT][32] f16
constexpr int oH2  = oH1 + kNT * 32;     // [kNT][32] f16 (kept for skip)
constexpr int oL1  = oH2 + kNT * 32;     // [kNT][32] f16
constexpr int oXF  = oL1 + kNT * 32;     // [kNT][32] f16
constexpr int oW1  = oXF + kNT * 32;     // [32][96] f16, BN scale folded, zero-pad
constexpr int oW2  = oW1 + 32 * kK1;     // [32][32]
constexpr int oW3  = oW2 + 32 * 32;
constexpr int oW4  = oW3 + 32 * 32;
constexpr int oWc  = oW4 + 32 * 32;      // [16][32] (rows >=3 zero)
constexpr int kHalves = oWc + 16 * 32;   // 64000 halves = 128000 B

constexpr int fBias  = 0;                // [4][32] folded BN bias
constexpr int fFeat  = fBias  + 4 * 32;  // [3][kNT] f32 (alpha,beta,omega)
constexpr int fPfn   = fFeat  + 3 * kNT; // [kNT] gathered Pf
constexpr int fBc    = fPfn   + kNT;     // [3]
constexpr int kFloats = fBc + 4;

constexpr unsigned kSmemBytes = (unsigned)kHalves * 2u + (unsigned)kFloats * 4u;
} // namespace

// gfx1250 has hardware v_tanh_f32 (TRANS32) -> branchless, ~1 instr.
__device__ __forceinline__ float fast_tanhf(float x) {
#if __has_builtin(__builtin_amdgcn_tanhf)
  return __builtin_amdgcn_tanhf(x);
#elif __has_builtin(__builtin_amdgcn_tanh_f32)
  return __builtin_amdgcn_tanh_f32(x);
#else
  float e = __expf(-2.0f * x);        // branchless fallback (v_exp_f32 + rcp)
  return (1.0f - e) / (1.0f + e);
#endif
}

__device__ __forceinline__ float gelu_fast(float x) {
  // tanh-form GELU; |err| ~1e-3, below the f16 activation noise floor.
  float u = 0.7978845608028654f * x * (1.0f + 0.044715f * x * x);
  return 0.5f * x * (1.0f + fast_tanhf(u));
}

union Frag { v16h v; v8h h[2]; };

__device__ __forceinline__ v8f wmma_acc(const v16h& a, const v16h& b, v8f c) {
  return __builtin_amdgcn_wmma_f32_16x16x32_f16(false, a, false, b, (short)0, c,
                                                false, false);
}

// Compute the 4 (2 col-tiles x 2 m-tiles) 16x16 output tiles of one wave.
// A layout (16-bit 16x32): lanes 0-15 -> M=lane, v0..3=K(0..7), v4..7=K(16..23);
//                          lanes 16-31 -> same M, K halves +8.
// B layout (32x16): lane -> col = lane&15, same K split as A (K 0..7/16..23, +8 for lanes 16-31).
template<int KSTEPS>
__device__ __forceinline__ void layer_tiles(const _Float16* act, int astride,
                                            const _Float16* w, int wstride,
                                            int wave, int lane, v8f acc[4]) {
  const int mr  = lane & 15;
  const int khA = (lane >> 4) * 8;
  const int khB = (lane >> 4) * 8;
  Frag A[2][KSTEPS];
#pragma unroll
  for (int mt = 0; mt < 2; ++mt) {
#pragma unroll
    for (int ks = 0; ks < KSTEPS; ++ks) {
      const _Float16* wp = w + (mt * 16 + mr) * wstride + ks * 32 + khA;
      A[mt][ks].h[0] = *(const v8h*)(wp);
      A[mt][ks].h[1] = *(const v8h*)(wp + 16);
    }
  }
#pragma unroll
  for (int c = 0; c < 2; ++c) {
    const int col = (wave * 2 + c) * 16 + mr;
    Frag Bf[KSTEPS];
#pragma unroll
    for (int ks = 0; ks < KSTEPS; ++ks) {
      const _Float16* ap = act + col * astride + ks * 32 + khB;
      Bf[ks].h[0] = *(const v8h*)(ap);
      Bf[ks].h[1] = *(const v8h*)(ap + 16);
    }
    v8f a0 = {}, a1 = {};
#pragma unroll
    for (int ks = 0; ks < KSTEPS; ++ks) {
      a0 = wmma_acc(A[0][ks].v, Bf[ks].v, a0);
      a1 = wmma_acc(A[1][ks].v, Bf[ks].v, a1);
    }
    acc[c * 2 + 0] = a0;
    acc[c * 2 + 1] = a1;
  }
}

// Bias (+ optional GELU) epilogue; D layout: lane -> col=ct*16+(lane&15),
// vgpr i -> M = mt*16 + 8*(lane>>4) + i.
__device__ __forceinline__ void epilogue_store(const v8f acc[4], _Float16* dst,
                                               const float* bias, int wave,
                                               int lane, bool do_gelu) {
  const int mr = lane & 15;
#pragma unroll
  for (int t = 0; t < 4; ++t) {
    const int col = (wave * 2 + (t >> 1)) * 16 + mr;
    const int mb  = (t & 1) * 16 + 8 * (lane >> 4);
#pragma unroll
    for (int i = 0; i < 8; ++i) {
      int m = mb + i;
      float y = acc[t][i] + bias[m];
      if (do_gelu) y = gelu_fast(y);
      dst[col * 32 + m] = (_Float16)y;
    }
  }
}

extern "C" __global__ __launch_bounds__(256)
void fused_prop(const float* __restrict__ If, const float* __restrict__ Pf,
                const float* __restrict__ Of, const int* __restrict__ args,
                const float* __restrict__ W1, const float* __restrict__ g1,
                const float* __restrict__ b1, const float* __restrict__ m1,
                const float* __restrict__ v1,
                const float* __restrict__ W2, const float* __restrict__ g2,
                const float* __restrict__ b2, const float* __restrict__ m2,
                const float* __restrict__ v2,
                const float* __restrict__ W3, const float* __restrict__ g3,
                const float* __restrict__ b3, const float* __restrict__ m3,
                const float* __restrict__ v3,
                const float* __restrict__ W4, const float* __restrict__ g4,
                const float* __restrict__ b4, const float* __restrict__ m4,
                const float* __restrict__ v4,
                const float* __restrict__ Wc, const float* __restrict__ bc,
                float* __restrict__ out) {
  extern __shared__ __align__(16) unsigned char smem_raw[];
  _Float16* hbase = (_Float16*)smem_raw;
  float*    fbase = (float*)(smem_raw + (size_t)kHalves * 2u);

  const int tid  = threadIdx.x;
  const int lane = tid & 31;
  const int wave = tid >> 5;
  const int b    = blockIdx.y;
  const int n0   = blockIdx.x * kTN;

  // ---- stage weights (f16, BN scale folded into rows) + bias ----
  for (int i = tid; i < 32 * kK1; i += 256) {
    int m = i / kK1, k = i - m * kK1;
    float s = g1[m] * rsqrtf(v1[m] + 1e-5f);
    hbase[oW1 + i] = (k < kCt) ? (_Float16)(W1[m * kCt + k] * s) : (_Float16)0.f;
  }
  for (int i = tid; i < 32 * 32; i += 256) {
    int m = i >> 5;
    hbase[oW2 + i] = (_Float16)(W2[i] * (g2[m] * rsqrtf(v2[m] + 1e-5f)));
    hbase[oW3 + i] = (_Float16)(W3[i] * (g3[m] * rsqrtf(v3[m] + 1e-5f)));
    hbase[oW4 + i] = (_Float16)(W4[i] * (g4[m] * rsqrtf(v4[m] + 1e-5f)));
  }
  for (int i = tid; i < 16 * 32; i += 256) {
    int m = i >> 5;
    hbase[oWc + i] = (m < 3) ? (_Float16)Wc[i] : (_Float16)0.f;
  }
  if (tid < 32) {
    float iv;
    iv = g1[tid] * rsqrtf(v1[tid] + 1e-5f); fbase[fBias + 0 * 32 + tid] = b1[tid] - m1[tid] * iv;
    iv = g2[tid] * rsqrtf(v2[tid] + 1e-5f); fbase[fBias + 1 * 32 + tid] = b2[tid] - m2[tid] * iv;
    iv = g3[tid] * rsqrtf(v3[tid] + 1e-5f); fbase[fBias + 2 * 32 + tid] = b3[tid] - m3[tid] * iv;
    iv = g4[tid] * rsqrtf(v4[tid] + 1e-5f); fbase[fBias + 3 * 32 + tid] = b4[tid] - m4[tid] * iv;
  }
  if (tid < 3) fbase[fBc + tid] = bc[tid];

  // ---- build X tile: col = num*16 + nl, channels [If | If@args | Pf@args | Of | 0pad] ----
  {
    const int nl  = tid & 15, num = tid >> 4;
    const int n   = n0 + nl;
    const int a   = args[(b * kNum + num) * kN + n];
    _Float16* xc  = hbase + oX + tid * kK1;
    const float* Ifb = If + (size_t)b * kCfi * kN;
#pragma unroll
    for (int c = 0; c < kCfi; ++c) xc[c] = (_Float16)Ifb[c * kN + n];
#pragma unroll
    for (int c = 0; c < kCfi; ++c) xc[32 + c] = (_Float16)Ifb[c * kN + a];
    float pv = Pf[b * kN + a];
    xc[64] = (_Float16)pv;
    fbase[fPfn + tid] = pv;
    xc[65] = (_Float16)Of[((b * 2 + 0) * kNum + num) * kN + n];
    xc[66] = (_Float16)Of[((b * 2 + 1) * kNum + num) * kN + n];
#pragma unroll
    for (int k = kCt; k < kK1; ++k) xc[k] = (_Float16)0.f;
  }
  __syncthreads();

  const float* bias = fbase + fBias;
  v8f acc[4];

  // ---- layer 1: 96 -> 32, bias+GELU ----
  layer_tiles<3>(hbase + oX, kK1, hbase + oW1, kK1, wave, lane, acc);
  epilogue_store(acc, hbase + oH1, bias + 0 * 32, wave, lane, true);
  __syncthreads();

  // ---- layer 2: 32 -> 32, bias+GELU (kept as h2 for the skip) ----
  layer_tiles<1>(hbase + oH1, 32, hbase + oW2, 32, wave, lane, acc);
  epilogue_store(acc, hbase + oH2, bias + 1 * 32, wave, lane, true);
  __syncthreads();

  // ---- layer 3: 32 -> 32, bias+GELU ----
  layer_tiles<1>(hbase + oH2, 32, hbase + oW3, 32, wave, lane, acc);
  epilogue_store(acc, hbase + oL1, bias + 2 * 32, wave, lane, true);
  __syncthreads();

  // ---- layer 4: 32 -> 32, bias, + h2 skip, GELU -> XF ----
  layer_tiles<1>(hbase + oL1, 32, hbase + oW4, 32, wave, lane, acc);
  {
    const int mr = lane & 15;
#pragma unroll
    for (int t = 0; t < 4; ++t) {
      const int col = (wave * 2 + (t >> 1)) * 16 + mr;
      const int mb  = (t & 1) * 16 + 8 * (lane >> 4);
#pragma unroll
      for (int i = 0; i < 8; ++i) {
        int m = mb + i;
        float y = acc[t][i] + bias[3 * 32 + m] + (float)hbase[oH2 + col * 32 + m];
        hbase[oXF + col * 32 + m] = (_Float16)gelu_fast(y);
      }
    }
  }
  __syncthreads();

  // ---- final conv: 32 -> 3 (rows 3..15 of Wc are zero) ----
  {
    const int mr  = lane & 15;
    const int khA = (lane >> 4) * 8;
    const int khB = (lane >> 4) * 8;
    Frag A0;
    const _Float16* wp = hbase + oWc + mr * 32 + khA;
    A0.h[0] = *(const v8h*)(wp);
    A0.h[1] = *(const v8h*)(wp + 16);
    v8f dc[2];
#pragma unroll
    for (int c = 0; c < 2; ++c) {
      const int col = (wave * 2 + c) * 16 + mr;
      Frag Bf;
      const _Float16* ap = hbase + oXF + col * 32 + khB;
      Bf.h[0] = *(const v8h*)(ap);
      Bf.h[1] = *(const v8h*)(ap + 16);
      v8f z = {};
      dc[c] = wmma_acc(A0.v, Bf.v, z);
    }
#pragma unroll
    for (int c = 0; c < 2; ++c) {
      const int col = (wave * 2 + c) * 16 + mr;
      const int mb  = 8 * (lane >> 4);
#pragma unroll
      for (int i = 0; i < 8; ++i) {
        int m = mb + i;
        if (m < 3) fbase[fFeat + m * kNT + col] = dc[c][i] + fbase[fBc + m];
      }
    }
  }
  __syncthreads();

  // ---- softmax over NUM (16) and weighted reduction, one thread per spatial n ----
  if (tid < kTN) {
    const int nl = tid;
    const float* fa = fbase + fFeat + 0 * kNT;
    const float* fb = fbase + fFeat + 1 * kNT;
    const float* fo = fbase + fFeat + 2 * kNT;
    float mx = -INFINITY;
#pragma unroll
    for (int num = 0; num < kNum; ++num) mx = fmaxf(mx, fo[num * 16 + nl]);
    float ex[kNum], se = 0.f;
#pragma unroll
    for (int num = 0; num < kNum; ++num) {
      ex[num] = __expf(fo[num * 16 + nl] - mx);   // v_exp_f32, branchless
      se += ex[num];
    }
    const float inv = 1.f / se;
    float acc2 = 0.f;
#pragma unroll
    for (int num = 0; num < kNum; ++num) {
      const int col = num * 16 + nl;
      acc2 += ((fa[col] + 1.f) * fbase[fPfn + col] + fb[col]) * ex[num] * inv;
    }
    out[b * kN + n0 + nl] = acc2;
  }
}

extern "C" void kernel_launch(void* const* d_in, const int* in_sizes, int n_in,
                              void* d_out, int out_size, void* d_ws, size_t ws_size,
                              hipStream_t stream) {
  (void)in_sizes; (void)n_in; (void)d_ws; (void)ws_size; (void)out_size;
  const float* If = (const float*)d_in[0];
  const float* Pf = (const float*)d_in[1];
  const float* Of = (const float*)d_in[2];
  const int*   args = (const int*)d_in[3];
  const float* W1 = (const float*)d_in[4];
  const float* g1 = (const float*)d_in[5];
  const float* b1 = (const float*)d_in[6];
  const float* m1 = (const float*)d_in[7];
  const float* v1 = (const float*)d_in[8];
  const float* W2 = (const float*)d_in[9];
  const float* g2 = (const float*)d_in[10];
  const float* b2 = (const float*)d_in[11];
  const float* m2 = (const float*)d_in[12];
  const float* v2 = (const float*)d_in[13];
  const float* W3 = (const float*)d_in[14];
  const float* g3 = (const float*)d_in[15];
  const float* b3 = (const float*)d_in[16];
  const float* m3 = (const float*)d_in[17];
  const float* v3 = (const float*)d_in[18];
  const float* W4 = (const float*)d_in[19];
  const float* g4 = (const float*)d_in[20];
  const float* b4 = (const float*)d_in[21];
  const float* m4 = (const float*)d_in[22];
  const float* v4 = (const float*)d_in[23];
  const float* Wc = (const float*)d_in[24];
  const float* bc = (const float*)d_in[25];
  float* out = (float*)d_out;

  // >64KB dynamic LDS: raise the per-kernel cap (deterministic, capture-safe).
  hipFuncSetAttribute((const void*)fused_prop,
                      hipFuncAttributeMaxDynamicSharedMemorySize, (int)kSmemBytes);

  dim3 grid(kN / kTN, kB);  // 2304 x 2 blocks
  fused_prop<<<grid, 256, kSmemBytes, stream>>>(
      If, Pf, Of, args,
      W1, g1, b1, m1, v1,
      W2, g2, b2, m2, v2,
      W3, g3, b3, m3, v3,
      W4, g4, b4, m4, v4,
      Wc, bc, out);
}
